// EncoderLayer_24764781429513
// MI455X (gfx1250) — matrix operand order repaired and checked by reading, not compiled, not run on hardware
//
#include <hip/hip_runtime.h>
#include <hip/hip_bf16.h>
#include <stdint.h>

// ---------------------------------------------------------------------------
// Encoder layer for MI455X (gfx1250), wave32 + WMMA bf16 (16x16x32, f32 acc).
//   transpose-cvt weights->bf16 [N][K] | LN1 | 3x GEMM (Q,K,V) | V transpose |
//   attention (WMMA, clamp-based fixed-max softmax, rowsum via WMMA) |
//   GEMM+res (Wo) | LN2 | GEMM+ReLU (W1) | GEMM+ReLU+res (W2 -> out f32)
// All GEMM/attention tile fills are async global->LDS DMA (ASYNCcnt),
// double-buffered so the DMA overlaps the WMMAs.
// ---------------------------------------------------------------------------

typedef __bf16 bf16;
typedef __attribute__((ext_vector_type(16))) __bf16 v16bf;
typedef __attribute__((ext_vector_type(8)))  float  v8f;

union Frag { uint4 u[2]; v16bf v; };

__device__ __forceinline__ v8f wmma_bf16(v16bf a, v16bf b, v8f c) {
  return __builtin_amdgcn_wmma_f32_16x16x32_bf16(false, a, false, b, (short)0, c,
                                                 false, false);
}

// async copy: 16B per lane (or 32B with the x2 form), global -> LDS
__device__ __forceinline__ void async_b128(unsigned lds_off,
                                           unsigned long long gaddr) {
  asm volatile("global_load_async_to_lds_b128 %0, %1, off"
               :: "v"(lds_off), "v"(gaddr) : "memory");
}
__device__ __forceinline__ void async_b128x2(unsigned lds_off,
                                             unsigned long long gaddr) {
  asm volatile("global_load_async_to_lds_b128 %0, %1, off\n\t"
               "global_load_async_to_lds_b128 %0, %1, off offset:16"
               :: "v"(lds_off), "v"(gaddr) : "memory");
}
__device__ __forceinline__ void wait_async0() {
  asm volatile("s_wait_asynccnt 0x0" ::: "memory");
}
__device__ __forceinline__ void wait_ds0() {
  asm volatile("s_wait_dscnt 0x0" ::: "memory");
}

constexpr int Bn  = 8;
constexpr int Sn  = 1024;
constexpr int Dn  = 1024;
constexpr int Hn  = 16;
constexpr int DKn = 64;
constexpr int DHn = 2048;
constexpr int Mrows = Bn * Sn;   // 8192

// ---------------------------------------------------------------------------
// f32 [K][N] -> bf16 [N][K] tiled transpose-convert (weights, done once)
// ---------------------------------------------------------------------------
__global__ __launch_bounds__(256)
void cvt_transpose_kernel(const float* __restrict__ src, bf16* __restrict__ dst,
                          int K, int N) {
  __shared__ float tile[32][33];
  const int n0 = blockIdx.x * 32;
  const int k0 = blockIdx.y * 32;
  const int t  = threadIdx.x;
  const int rl = t >> 3;         // 0..31
  const int c4 = (t & 7) << 2;   // 0..28
  const float4 v = *(const float4*)(src + (size_t)(k0 + rl) * N + n0 + c4);
  tile[rl][c4 + 0] = v.x; tile[rl][c4 + 1] = v.y;
  tile[rl][c4 + 2] = v.z; tile[rl][c4 + 3] = v.w;
  __syncthreads();
  bf16* d = dst + (size_t)(n0 + rl) * K + k0 + c4;
  d[0] = (bf16)tile[c4 + 0][rl];
  d[1] = (bf16)tile[c4 + 1][rl];
  d[2] = (bf16)tile[c4 + 2][rl];
  d[3] = (bf16)tile[c4 + 3][rl];
}

// ---------------------------------------------------------------------------
// bf16 V transpose: [B,S,H,DK] -> Vt[b*H+h][DK][S]  (done once per layer)
// ---------------------------------------------------------------------------
__global__ __launch_bounds__(256)
void v_transpose_kernel(const bf16* __restrict__ Vb, bf16* __restrict__ Vt) {
  __shared__ bf16 tile[32][36];
  const int s0 = blockIdx.x * 32;
  const int d0 = blockIdx.y * 32;
  const int bh = blockIdx.z;
  const int b  = bh / Hn, h = bh % Hn;
  const int t  = threadIdx.x;
  const int rl = t >> 3;
  const int c4 = (t & 7) << 2;
  const bf16* sp = Vb + ((size_t)b * Sn + s0 + rl) * Dn + h * DKn + d0 + c4;
  tile[rl][c4 + 0] = sp[0]; tile[rl][c4 + 1] = sp[1];
  tile[rl][c4 + 2] = sp[2]; tile[rl][c4 + 3] = sp[3];
  __syncthreads();
  bf16* dp = Vt + ((size_t)bh * DKn + d0 + rl) * Sn + s0 + c4;
  dp[0] = tile[c4 + 0][rl]; dp[1] = tile[c4 + 1][rl];
  dp[2] = tile[c4 + 2][rl]; dp[3] = tile[c4 + 3][rl];
}

// ---------------------------------------------------------------------------
__global__ __launch_bounds__(256)
void ln_kernel(const float* __restrict__ X, const float* __restrict__ g,
               const float* __restrict__ b, float* __restrict__ Yf,
               bf16* __restrict__ Yb) {
  const int row  = blockIdx.x;
  const int t    = threadIdx.x;
  const int lane = t & 31;
  const int wave = t >> 5;

  const float4 v = ((const float4*)(X + (size_t)row * Dn))[t];
  float s  = v.x + v.y + v.z + v.w;
  float s2 = v.x * v.x + v.y * v.y + v.z * v.z + v.w * v.w;
#pragma unroll
  for (int off = 1; off < 32; off <<= 1) {
    s  += __shfl_xor(s,  off, 32);
    s2 += __shfl_xor(s2, off, 32);
  }
  __shared__ float rs[8], rs2[8];
  if (lane == 0) { rs[wave] = s; rs2[wave] = s2; }
  __syncthreads();
  float ts = 0.f, ts2 = 0.f;
#pragma unroll
  for (int i = 0; i < 8; ++i) { ts += rs[i]; ts2 += rs2[i]; }
  const float mu   = ts * (1.0f / Dn);
  const float var  = ts2 * (1.0f / Dn) - mu * mu;
  const float rstd = rsqrtf(var + 1e-5f);

  const float4 gg = ((const float4*)g)[t];
  const float4 bb = ((const float4*)b)[t];
  float4 y;
  y.x = (v.x - mu) * rstd * gg.x + bb.x;
  y.y = (v.y - mu) * rstd * gg.y + bb.y;
  y.z = (v.z - mu) * rstd * gg.z + bb.z;
  y.w = (v.w - mu) * rstd * gg.w + bb.w;
  ((float4*)(Yf + (size_t)row * Dn))[t] = y;
  bf16* yb = Yb + (size_t)row * Dn + 4 * t;
  yb[0] = (bf16)y.x; yb[1] = (bf16)y.y; yb[2] = (bf16)y.z; yb[3] = (bf16)y.w;
}

// ---------------------------------------------------------------------------
// GEMM: C[M,N] = A[M,K] bf16 @ Wt[N,K] bf16 (weights pre-transposed), f32 acc.
// 256 thr = 8 waves (4 in M x 2 in N); block tile 128x128; wave tile 32x64.
// Both tiles async-DMA'd into double-buffered LDS; DMA overlaps WMMAs.
// EPI: 0 bf16 | 1 +res f32 | 2 relu bf16 | 3 res+relu f32
// ---------------------------------------------------------------------------
constexpr int BM = 128, BNB = 128, BK = 32, LPAD = 8;

template <int EPI>
__global__ __launch_bounds__(256)
void gemm_bf16_kernel(const bf16* __restrict__ A, const bf16* __restrict__ Wt,
                      const float* __restrict__ Res, float* __restrict__ Cf,
                      bf16* __restrict__ Cb, int N, int K) {
  __shared__ bf16 As[2][BM][BK + LPAD];    // [m][k]
  __shared__ bf16 Bs[2][BNB][BK + LPAD];   // [n][k]

  const int t    = threadIdx.x;
  const int lane = t & 31;
  const int wave = t >> 5;
  const int wy   = wave >> 1;              // 0..3 (M)
  const int wx   = wave & 1;               // 0..1 (N)
  const int m0   = blockIdx.y * BM;
  const int n0   = blockIdx.x * BNB;

  const int row16 = lane & 15;
  const int half  = lane >> 4;
  const int kqA   = half << 3;             // A frag: 0 or 8
  const int kb16  = half << 4;             // unused: 0 or 16; B frag also uses kqA

  const int rfill = t >> 1;                // 0..127
  const int kfill = (t & 1) << 4;          // 0 or 16

  v8f acc[2][4] = {};

  auto issue = [&](int k0, int buf) {
    async_b128x2((unsigned)(uintptr_t)&As[buf][rfill][kfill],
                 (unsigned long long)(uintptr_t)(
                     A + (size_t)(m0 + rfill) * K + k0 + kfill));
    async_b128x2((unsigned)(uintptr_t)&Bs[buf][rfill][kfill],
                 (unsigned long long)(uintptr_t)(
                     Wt + (size_t)(n0 + rfill) * K + k0 + kfill));
  };

  issue(0, 0);
  wait_async0();
  __syncthreads();

  const int steps = K / BK;
  for (int it = 0; it < steps; ++it) {
    const int cur = it & 1;
    if (it + 1 < steps) issue((it + 1) * BK, cur ^ 1);

    Frag af[2], bf[4];
#pragma unroll
    for (int mi = 0; mi < 2; ++mi) {
      const int r = wy * 32 + mi * 16 + row16;
      af[mi].u[0] = *(const uint4*)&As[cur][r][kqA];
      af[mi].u[1] = *(const uint4*)&As[cur][r][16 + kqA];
    }
#pragma unroll
    for (int ni = 0; ni < 4; ++ni) {
      const int c = wx * 64 + ni * 16 + row16;
      bf[ni].u[0] = *(const uint4*)&Bs[cur][c][kqA];
      bf[ni].u[1] = *(const uint4*)&Bs[cur][c][16 + kqA];
    }
#pragma unroll
    for (int mi = 0; mi < 2; ++mi)
#pragma unroll
      for (int ni = 0; ni < 4; ++ni)
        acc[mi][ni] = wmma_bf16(af[mi].v, bf[ni].v, acc[mi][ni]);

    wait_async0();
    __syncthreads();
  }

#pragma unroll
  for (int mi = 0; mi < 2; ++mi)
#pragma unroll
    for (int ni = 0; ni < 4; ++ni)
#pragma unroll
      for (int r = 0; r < 8; ++r) {
        const int grow = m0 + wy * 32 + mi * 16 + half * 8 + r;
        const int gcol = n0 + wx * 64 + ni * 16 + row16;
        const size_t idx = (size_t)grow * N + gcol;
        float x = acc[mi][ni][r];
        if constexpr (EPI == 0) {
          Cb[idx] = (bf16)x;
        } else if constexpr (EPI == 1) {
          Cf[idx] = x + Res[idx];
        } else if constexpr (EPI == 2) {
          Cb[idx] = (bf16)fmaxf(x, 0.0f);
        } else {
          Cf[idx] = Res[idx] + fmaxf(x, 0.0f);
        }
      }
}

// ---------------------------------------------------------------------------
// Attention: grid (S/128, B*H); 8 waves/block, 16 q-rows per wave.
// Clamp to [-5,5] => fixed-max softmax (exact after normalization); row sums
// via an extra WMMA against an all-ones B fragment. K and (pre-transposed) V
// tiles async-DMA'd into double-buffered LDS.
// ---------------------------------------------------------------------------
__global__ __launch_bounds__(256)
void attn_kernel(const bf16* __restrict__ Q, const bf16* __restrict__ Km,
                 const bf16* __restrict__ Vt, bf16* __restrict__ Ctx) {
  __shared__ bf16 Klds[2][32][64 + LPAD];    // [key][d]
  __shared__ bf16 Vlds[2][64][32 + LPAD];    // [d][key]
  __shared__ bf16 Plds[8][16][32 + LPAD];    // per-wave P tile [qrow][key]

  const int t    = threadIdx.x;
  const int lane = t & 31;
  const int wave = t >> 5;
  const int bh   = blockIdx.y;
  const int bidx = bh / Hn;
  const int h    = bh % Hn;
  const size_t base  = (size_t)bidx * Sn * Dn + (size_t)h * DKn;
  const size_t vbase = (size_t)bh * DKn * Sn;

  const int row16 = lane & 15;
  const int half  = lane >> 4;
  const int kqA   = half << 3;
  const int kb16  = half << 4;
  const int qrow0 = blockIdx.x * 128 + wave * 16;

  Frag qf[2];
  {
    const bf16* qp = Q + base + (size_t)(qrow0 + row16) * Dn;
    qf[0].u[0] = *(const uint4*)(qp + kqA);
    qf[0].u[1] = *(const uint4*)(qp + 16 + kqA);
    qf[1].u[0] = *(const uint4*)(qp + 32 + kqA);
    qf[1].u[1] = *(const uint4*)(qp + 48 + kqA);
  }

  v16bf ones;
#pragma unroll
  for (int i = 0; i < 16; ++i) ones[i] = (bf16)1.0f;

  v8f acc[4] = {};
  v8f lacc   = {};
  const float scale = 0.125f;  // DK^-0.5

  auto issue_kv = [&](int s0, int buf) {
    {  // K tile [32][64]
      const int r  = t >> 3;
      const int d8 = (t & 7) << 3;
      async_b128((unsigned)(uintptr_t)&Klds[buf][r][d8],
                 (unsigned long long)(uintptr_t)(
                     Km + base + (size_t)(s0 + r) * Dn + d8));
    }
    {  // V tile [64][32] from pre-transposed Vt
      const int d  = t >> 2;
      const int s8 = (t & 3) << 3;
      async_b128((unsigned)(uintptr_t)&Vlds[buf][d][s8],
                 (unsigned long long)(uintptr_t)(
                     Vt + vbase + (size_t)d * Sn + s0 + s8));
    }
  };

  issue_kv(0, 0);
  wait_async0();
  __syncthreads();

  for (int s0 = 0; s0 < Sn; s0 += 32) {
    const int cur = (s0 >> 5) & 1;
    if (s0 + 32 < Sn) issue_kv(s0 + 32, cur ^ 1);

    // scores for key cols [s0, s0+16) and [s0+16, s0+32)
    v8f sc[2] = {};
#pragma unroll
    for (int ct = 0; ct < 2; ++ct) {
      const int kr = ct * 16 + row16;
      Frag kf;
      kf.u[0] = *(const uint4*)&Klds[cur][kr][kqA];
      kf.u[1] = *(const uint4*)&Klds[cur][kr][16 + kqA];
      sc[ct] = wmma_bf16(qf[0].v, kf.v, sc[ct]);
      kf.u[0] = *(const uint4*)&Klds[cur][kr][32 + kqA];
      kf.u[1] = *(const uint4*)&Klds[cur][kr][48 + kqA];
      sc[ct] = wmma_bf16(qf[1].v, kf.v, sc[ct]);
    }

    // clamp(-5,5) + exp(s - 5)  (fixed-max softmax; exact after normalize)
#pragma unroll
    for (int r = 0; r < 8; ++r) {
      const float v0 = fminf(fmaxf(sc[0][r] * scale, -5.0f), 5.0f);
      const float v1 = fminf(fmaxf(sc[1][r] * scale, -5.0f), 5.0f);
      const float p0 = __expf(v0 - 5.0f);
      const float p1 = __expf(v1 - 5.0f);
      const int m = half * 8 + r;
      Plds[wave][m][row16]      = (bf16)p0;
      Plds[wave][m][16 + row16] = (bf16)p1;
    }
    wait_ds0();   // wave-private tile: enforce LDS store->load order

    // P @ V (4 d-tiles) + row-sum via ones column
    Frag pf;
    pf.u[0] = *(const uint4*)&Plds[wave][row16][kqA];
    pf.u[1] = *(const uint4*)&Plds[wave][row16][16 + kqA];
#pragma unroll
    for (int nt = 0; nt < 4; ++nt) {
      const int d = nt * 16 + row16;
      Frag vf;
      vf.u[0] = *(const uint4*)&Vlds[cur][d][kqA];
      vf.u[1] = *(const uint4*)&Vlds[cur][d][16 + kqA];
      acc[nt] = wmma_bf16(pf.v, vf.v, acc[nt]);
    }
    lacc = wmma_bf16(pf.v, ones, lacc);

    wait_async0();
    __syncthreads();
  }

  // ctx = acc / rowsum  ->  bf16 [B,S,H,DK] == [Mrows, D]
#pragma unroll
  for (int nt = 0; nt < 4; ++nt)
#pragma unroll
    for (int r = 0; r < 8; ++r) {
      const int m = half * 8 + r;
      const float o = acc[nt][r] / lacc[r];
      Ctx[base + (size_t)(qrow0 + m) * Dn + nt * 16 + row16] = (bf16)o;
    }
}

// ---------------------------------------------------------------------------
extern "C" void kernel_launch(void* const* d_in, const int* in_sizes, int n_in,
                              void* d_out, int out_size, void* d_ws,
                              size_t ws_size, hipStream_t stream) {
  const float* n_f  = (const float*)d_in[0];
  const float* Wq   = (const float*)d_in[1];
  const float* Wk   = (const float*)d_in[2];
  const float* Wv   = (const float*)d_in[3];
  const float* Wo   = (const float*)d_in[4];
  const float* W1   = (const float*)d_in[5];
  const float* W2   = (const float*)d_in[6];
  const float* ln1g = (const float*)d_in[7];
  const float* ln1b = (const float*)d_in[8];
  const float* ln2g = (const float*)d_in[9];
  const float* ln2b = (const float*)d_in[10];
  float* out = (float*)d_out;

  char* ws = (char*)d_ws;
  size_t off = 0;
  auto alloc = [&](size_t bytes) -> void* {
    void* p = ws + off;
    off = (off + bytes + 255) & ~(size_t)255;
    return p;
  };

  bf16* wq_t   = (bf16*)alloc((size_t)Dn * Dn * 2);    // [N][K]
  bf16* wk_t   = (bf16*)alloc((size_t)Dn * Dn * 2);
  bf16* wv_t   = (bf16*)alloc((size_t)Dn * Dn * 2);
  bf16* wo_t   = (bf16*)alloc((size_t)Dn * Dn * 2);
  bf16* w1_t   = (bf16*)alloc((size_t)Dn * DHn * 2);   // [DH][D]
  bf16* w2_t   = (bf16*)alloc((size_t)DHn * Dn * 2);   // [D][DH]
  float* x_f   = (float*)alloc((size_t)Mrows * Dn * 4);
  bf16* x_b    = (bf16*)alloc((size_t)Mrows * Dn * 2);
  bf16* q_b    = (bf16*)alloc((size_t)Mrows * Dn * 2);
  bf16* k_b    = (bf16*)alloc((size_t)Mrows * Dn * 2);
  bf16* v_b    = (bf16*)alloc((size_t)Mrows * Dn * 2);
  bf16* vt_b   = (bf16*)alloc((size_t)Mrows * Dn * 2); // [B*H][DK][S]
  bf16* ctx_b  = (bf16*)alloc((size_t)Mrows * Dn * 2);
  float* xat_f = (float*)alloc((size_t)Mrows * Dn * 4);
  float* x2_f  = (float*)alloc((size_t)Mrows * Dn * 4);
  bf16* x2_b   = (bf16*)alloc((size_t)Mrows * Dn * 2);
  bf16* h_b    = (bf16*)alloc((size_t)Mrows * DHn * 2);
  (void)ws_size; (void)n_in; (void)in_sizes; (void)out_size;

  // weights: transpose-convert to bf16 [N][K]
  {
    dim3 g1(Dn / 32, Dn / 32);
    cvt_transpose_kernel<<<g1, 256, 0, stream>>>(Wq, wq_t, Dn, Dn);
    cvt_transpose_kernel<<<g1, 256, 0, stream>>>(Wk, wk_t, Dn, Dn);
    cvt_transpose_kernel<<<g1, 256, 0, stream>>>(Wv, wv_t, Dn, Dn);
    cvt_transpose_kernel<<<g1, 256, 0, stream>>>(Wo, wo_t, Dn, Dn);
    dim3 g2(DHn / 32, Dn / 32);   // W1: [D][DH] -> [DH][D]
    cvt_transpose_kernel<<<g2, 256, 0, stream>>>(W1, w1_t, Dn, DHn);
    dim3 g3(Dn / 32, DHn / 32);   // W2: [DH][D] -> [D][DH]
    cvt_transpose_kernel<<<g3, 256, 0, stream>>>(W2, w2_t, DHn, Dn);
  }

  ln_kernel<<<Mrows, 256, 0, stream>>>(n_f, ln1g, ln1b, x_f, x_b);

  {
    dim3 grid(Dn / BNB, Mrows / BM);
    gemm_bf16_kernel<0><<<grid, 256, 0, stream>>>(x_b, wq_t, nullptr, nullptr,
                                                  q_b, Dn, Dn);
    gemm_bf16_kernel<0><<<grid, 256, 0, stream>>>(x_b, wk_t, nullptr, nullptr,
                                                  k_b, Dn, Dn);
    gemm_bf16_kernel<0><<<grid, 256, 0, stream>>>(x_b, wv_t, nullptr, nullptr,
                                                  v_b, Dn, Dn);
  }
  {
    dim3 grid(Sn / 32, DKn / 32, Bn * Hn);
    v_transpose_kernel<<<grid, 256, 0, stream>>>(v_b, vt_b);
  }
  {
    dim3 grid(Sn / 128, Bn * Hn);
    attn_kernel<<<grid, 256, 0, stream>>>(q_b, k_b, vt_b, ctx_b);
  }
  {
    dim3 grid(Dn / BNB, Mrows / BM);
    gemm_bf16_kernel<1><<<grid, 256, 0, stream>>>(ctx_b, wo_t, x_f, xat_f,
                                                  nullptr, Dn, Dn);
  }
  ln_kernel<<<Mrows, 256, 0, stream>>>(xat_f, ln2g, ln2b, x2_f, x2_b);
  {
    dim3 grid(DHn / BNB, Mrows / BM);
    gemm_bf16_kernel<2><<<grid, 256, 0, stream>>>(x2_b, w1_t, nullptr, nullptr,
                                                  h_b, DHn, Dn);
  }
  {
    dim3 grid(Dn / BNB, Mrows / BM);
    gemm_bf16_kernel<3><<<grid, 256, 0, stream>>>(h_b, w2_t, x2_f, out,
                                                  nullptr, Dn, DHn);
  }
}
